// GATv2Network_79860621902477
// MI455X (gfx1250) — matrix operand order repaired and checked by reading, not compiled, not run on hardware
//
#include <hip/hip_runtime.h>

#define NN   50000
#define EE   500000
#define INC  128
#define HIDC 64
#define OUTC 256   // heads * hidden

typedef __attribute__((ext_vector_type(16))) __bf16 v16bf;
typedef __attribute__((ext_vector_type(8)))  float  v8f;

union BFrag { uint4 u[2]; v16bf v; };

// ---------------- WMMA helpers ----------------

__device__ __forceinline__ v8f wmma_bf16(v16bf a, v16bf b, v8f c) {
  // v_wmma_f32_16x16x32_bf16
  return __builtin_amdgcn_wmma_f32_16x16x32_bf16(
      false, a, false, b, (short)0, c, false, false);
}

// A (16x32, bf16, row-major, leading dim = lda):
// lanes 0-15: row = lane,    K = k0 + {0..7, 16..23}
// lanes 16-31: row = lane-16, K = k0 + {8..15, 24..31}
__device__ __forceinline__ v16bf load_a_bf16(const __bf16* A, int row, int lda,
                                             int k0, int lane) {
  int base = k0 + ((lane & 16) ? 8 : 0);
  const uint4* p = (const uint4*)(A + row * lda + base);
  BFrag f;
  f.u[0] = p[0];   // 8 bf16: K = base .. base+7
  f.u[1] = p[2];   // 8 bf16: K = base+16 .. base+23
  return f.v;
}

// Same fragment pattern, but source row is f32 (converted to bf16 in regs).
__device__ __forceinline__ v16bf load_a_f32row(const float* rp, int k0, int lane) {
  int base = k0 + ((lane & 16) ? 8 : 0);
  const float4* q = (const float4*)(rp + base);
  float4 f0 = q[0], f1 = q[1], f2 = q[4], f3 = q[5];
  v16bf v;
  v[0] = (__bf16)f0.x;  v[1] = (__bf16)f0.y;  v[2] = (__bf16)f0.z;  v[3] = (__bf16)f0.w;
  v[4] = (__bf16)f1.x;  v[5] = (__bf16)f1.y;  v[6] = (__bf16)f1.z;  v[7] = (__bf16)f1.w;
  v[8] = (__bf16)f2.x;  v[9] = (__bf16)f2.y;  v[10] = (__bf16)f2.z; v[11] = (__bf16)f2.w;
  v[12] = (__bf16)f3.x; v[13] = (__bf16)f3.y; v[14] = (__bf16)f3.z; v[15] = (__bf16)f3.w;
  return v;
}

// B (32x16, bf16) sourced from LDS holding B^T row-major [Nout][K]:
// lanes 0-15: col n = lane,    K = k0 + {0..7, 16..23}
// lanes 16-31: col n = lane-16, K = k0 + {8..15, 24..31}
__device__ __forceinline__ v16bf load_b_lds(const __bf16* sB, int K, int nt,
                                            int k0, int lane) {
  int n   = nt * 16 + (lane & 15);
  int klo = k0 + ((lane & 16) ? 8 : 0);
  const uint4* p = (const uint4*)(sB + n * K + klo);
  BFrag f;
  f.u[0] = p[0];
  f.u[1] = p[2];
  return f.v;
}

// order-preserving float <-> u32 map for atomic max
__device__ __forceinline__ unsigned fmap(float f) {
  unsigned u = __float_as_uint(f);
  return (u & 0x80000000u) ? ~u : (u | 0x80000000u);
}
__device__ __forceinline__ float funmap(unsigned u) {
  unsigned b = (u & 0x80000000u) ? (u & 0x7fffffffu) : ~u;
  return __uint_as_float(b);
}

// ---------------- small utility kernels ----------------

__global__ __launch_bounds__(256) void k_tobf(const float* __restrict__ a,
                                              __bf16* __restrict__ b, int n) {
  int i = blockIdx.x * 256 + threadIdx.x;
  if (i < n) b[i] = (__bf16)a[i];
}

// W [K][Nout] f32 -> WT [Nout][K] bf16
__global__ __launch_bounds__(256) void k_wtrans(const float* __restrict__ W,
                                                __bf16* __restrict__ WT,
                                                int K, int Nout) {
  int i = blockIdx.x * 256 + threadIdx.x;
  if (i >= K * Nout) return;
  int k = i / Nout, n = i - k * Nout;
  WT[n * K + k] = (__bf16)W[i];
}

__global__ __launch_bounds__(256) void k_zero(unsigned* __restrict__ p, int n) {
  int i = blockIdx.x * 256 + threadIdx.x;
  if (i < n) p[i] = 0u;
}

__global__ __launch_bounds__(256) void k_copyf(const float* __restrict__ a,
                                               float* __restrict__ b, int n) {
  int i = blockIdx.x * 256 + threadIdx.x;
  if (i < n) b[i] = a[i];
}

// ---------------- node-side GEMM: C[M,NOUT] = A_bf16[M,K] @ B + bias ----------------
// B supplied transposed+bf16 as BT[NOUT][K]; staged in LDS.
// K, NOUT are compile-time so fragment arrays stay in fixed VGPRs (no v_movrels).

template <int K, int NOUT>
__global__ __launch_bounds__(256) void k_gemm_node(
    const __bf16* __restrict__ A, const __bf16* __restrict__ BT,
    const float* __restrict__ bias, float* __restrict__ C, int M) {
  __shared__ __bf16 sB[NOUT * K];
  int tid = threadIdx.x;
  {
    const int cnt = (K * NOUT) >> 3;
    const uint4* gB = (const uint4*)BT;
    uint4* sB4 = (uint4*)sB;
#pragma unroll 4
    for (int i = tid; i < cnt; i += 256) sB4[i] = gB[i];
  }
  __syncthreads();

  int wv = tid >> 5, lane = tid & 31;
  int mt = blockIdx.x * 8 + wv;
  if (mt * 16 >= M) return;

  constexpr int KK = K >> 5;
  v16bf aF[KK];
  int row = mt * 16 + (lane & 15);
#pragma unroll
  for (int k = 0; k < KK; ++k) aF[k] = load_a_bf16(A, row, K, k << 5, lane);

  constexpr int NT = NOUT >> 4;
  int coln = lane & 15;
  int m0   = mt * 16 + ((lane & 16) ? 8 : 0);
#pragma unroll
  for (int nt = 0; nt < NT; ++nt) {
    float bv = bias[nt * 16 + coln];
    v8f acc;
#pragma unroll
    for (int i = 0; i < 8; ++i) acc[i] = bv;
#pragma unroll
    for (int k = 0; k < KK; ++k) {
      v16bf b = load_b_lds(sB, K, nt, k << 5, lane);
      acc = wmma_bf16(aF[k], b, acc);
    }
    int col = nt * 16 + coln;
#pragma unroll
    for (int j = 0; j < 8; ++j) C[(m0 + j) * NOUT + col] = acc[j];
  }
}

// ---------------- fused edge kernel: GEMM + attention logits ----------------
// per 16-edge tile: ee = edge_attr @ We (WMMA, f32->bf16 on the fly),
// z = lrelu(ee + xl[src] + xr[dst]); logits[e][h] = sum_c z*att; atomic-max amax.

__global__ __launch_bounds__(256) void k_edge_logits(
    const float* __restrict__ ea, const __bf16* __restrict__ WeT,
    const float* __restrict__ xl, const float* __restrict__ xr,
    const float* __restrict__ att,
    const int* __restrict__ src, const int* __restrict__ dst,
    float* __restrict__ logits, unsigned* __restrict__ amaxu) {
  __shared__ __bf16 sB[OUTC * INC];  // 64 KB: WeT [256][128]
  int tid = threadIdx.x;
  {
    const uint4* gB = (const uint4*)WeT;
    uint4* sB4 = (uint4*)sB;
#pragma unroll 4
    for (int i = tid; i < (OUTC * INC) >> 3; i += 256) sB4[i] = gB[i];
  }
  __syncthreads();

  int wv = tid >> 5, lane = tid & 31;
  int e0 = blockIdx.x * 128 + wv * 16;
  if (e0 >= EE) return;

  if (lane == 0 && blockIdx.x * 128 + 128 < EE)
    __builtin_prefetch(ea + (size_t)(blockIdx.x * 128 + 128) * INC, 0, 1);

  // cache A fragments (16 edges x 128 K, converted to bf16)
  v16bf aF[4];
  {
    const float* rp = ea + (size_t)(e0 + (lane & 15)) * INC;
#pragma unroll
    for (int k = 0; k < 4; ++k) aF[k] = load_a_f32row(rp, k << 5, lane);
  }

  int coln = lane & 15;
  int off8 = (lane & 16) ? 8 : 0;

  int sIdx[8], dIdx[8];
#pragma unroll
  for (int j = 0; j < 8; ++j) {
    sIdx[j] = src[e0 + off8 + j];
    dIdx[j] = dst[e0 + off8 + j];
  }

  float sacc[8][4];
#pragma unroll
  for (int j = 0; j < 8; ++j)
#pragma unroll
    for (int h = 0; h < 4; ++h) sacc[j][h] = 0.f;

#pragma unroll
  for (int nt = 0; nt < 16; ++nt) {
    v8f acc;
#pragma unroll
    for (int i = 0; i < 8; ++i) acc[i] = 0.f;
#pragma unroll
    for (int k = 0; k < 4; ++k) {
      v16bf b = load_b_lds(sB, INC, nt, k << 5, lane);
      acc = wmma_bf16(aF[k], b, acc);
    }
    int n = nt * 16 + coln;
    float av = att[n];
    int h = nt >> 2;
#pragma unroll
    for (int j = 0; j < 8; ++j) {
      float w = acc[j] + xl[sIdx[j] * OUTC + n] + xr[dIdx[j] * OUTC + n];
      float z = (w > 0.f) ? w : 0.2f * w;
      sacc[j][h] += z * av;
    }
  }

  // reduce over the 16 lanes of each half-group (xor masks stay in-group)
#pragma unroll
  for (int j = 0; j < 8; ++j)
#pragma unroll
    for (int h = 0; h < 4; ++h) {
      float v = sacc[j][h];
      v += __shfl_xor(v, 1, 32);
      v += __shfl_xor(v, 2, 32);
      v += __shfl_xor(v, 4, 32);
      v += __shfl_xor(v, 8, 32);
      sacc[j][h] = v;
    }

  if (coln == 0) {
#pragma unroll
    for (int j = 0; j < 8; ++j) {
      int e = e0 + off8 + j;
#pragma unroll
      for (int h = 0; h < 4; ++h) {
        float v = sacc[j][h];
        logits[e * 4 + h] = v;
        atomicMax(&amaxu[dIdx[j] * 4 + h], fmap(v));
      }
    }
  }
}

// ---------------- softmax exp + denom ----------------

__global__ __launch_bounds__(256) void k_edge_exp(
    const float* __restrict__ logits, const unsigned* __restrict__ amaxu,
    const int* __restrict__ dst, float* __restrict__ exv,
    float* __restrict__ denom) {
  int idx = blockIdx.x * 256 + threadIdx.x;
  if (idx >= EE * 4) return;
  int e = idx >> 2, h = idx & 3;
  int d = dst[e];
  float am = funmap(amaxu[d * 4 + h]);
  float ex = __expf(logits[idx] - am);
  exv[idx] = ex;
  atomicAdd(&denom[d * 4 + h], ex);
}

// ---------------- message aggregation: out[dst] += sum_h xl[src]*alpha / H ----------------

__global__ __launch_bounds__(256) void k_edge_agg(
    const float* __restrict__ exv, const float* __restrict__ denom,
    const float* __restrict__ xl, const int* __restrict__ src,
    const int* __restrict__ dst, float* __restrict__ outacc) {
  int idx = blockIdx.x * 256 + threadIdx.x;
  if (idx >= EE * 64) return;
  int e = idx >> 6, c = idx & 63;
  int s = src[e], d = dst[e];
  float acc = 0.f;
#pragma unroll
  for (int h = 0; h < 4; ++h) {
    float al = exv[e * 4 + h] / (denom[d * 4 + h] + 1e-16f);
    acc += xl[s * OUTC + h * 64 + c] * al;
  }
  atomicAdd(&outacc[d * 64 + c], acc * 0.25f);
}

// ---------------- finalize: +bias, BN (eval), +residual, ReLU, bf16 copy ----------------

__global__ __launch_bounds__(256) void k_finalize(
    const float* __restrict__ outacc, const float* __restrict__ bias,
    const float* __restrict__ gamma, const float* __restrict__ beta,
    const float* __restrict__ rm, const float* __restrict__ rv,
    const float* __restrict__ res, float* __restrict__ hout,
    __bf16* __restrict__ hbf, float* __restrict__ finalOut, int mode) {
  int idx = blockIdx.x * 256 + threadIdx.x;
  if (idx >= NN * HIDC) return;
  int c = idx & 63;
  float v = outacc[idx] + bias[c];
  if (mode == 0) {
    v = (v - rm[c]) * rsqrtf(rv[c] + 1e-5f) * gamma[c] + beta[c];
    v += res[idx];
    v = fmaxf(v, 0.f);
    hout[idx] = v;
    hbf[idx] = (__bf16)v;
  } else {
    finalOut[idx] = v;
  }
}

// ---------------- host-side orchestration ----------------

#define CDIV(a, b) (((a) + (b) - 1) / (b))

extern "C" void kernel_launch(void* const* d_in, const int* in_sizes, int n_in,
                              void* d_out, int out_size, void* d_ws, size_t ws_size,
                              hipStream_t stream) {
  (void)in_sizes; (void)n_in; (void)out_size; (void)ws_size;

  const float* x      = (const float*)d_in[0];
  const int*   ei     = (const int*)d_in[1];
  const float* ea     = (const float*)d_in[2];
  const float* resW   = (const float*)d_in[3];
  const float* resb   = (const float*)d_in[4];
  const float* Wl0    = (const float*)d_in[5];
  const float* bl0    = (const float*)d_in[6];
  const float* Wr0    = (const float*)d_in[7];
  const float* br0    = (const float*)d_in[8];
  const float* We0    = (const float*)d_in[9];
  const float* att0   = (const float*)d_in[10];
  const float* bias0  = (const float*)d_in[11];
  const float* Wl12   = (const float*)d_in[12];
  const float* bl12   = (const float*)d_in[13];
  const float* Wr12   = (const float*)d_in[14];
  const float* br12   = (const float*)d_in[15];
  const float* We12   = (const float*)d_in[16];
  const float* att12  = (const float*)d_in[17];
  const float* bias12 = (const float*)d_in[18];
  const float* bng    = (const float*)d_in[19];
  const float* bnb    = (const float*)d_in[20];
  const float* bnm    = (const float*)d_in[21];
  const float* bnv    = (const float*)d_in[22];

  const int* src = ei;
  const int* dst = ei + EE;

  char* w = (char*)d_ws;
  size_t off = 0;
  auto grab = [&](size_t bytes) -> char* {
    char* p = w + off;
    off = (off + bytes + 255) & ~(size_t)255;
    return p;
  };

  __bf16* xbf  = (__bf16*)grab((size_t)NN * INC * 2);
  __bf16* wlT  = (__bf16*)grab((size_t)OUTC * INC * 2);
  __bf16* wrT  = (__bf16*)grab((size_t)OUTC * INC * 2);
  __bf16* weT  = (__bf16*)grab((size_t)OUTC * INC * 2);
  __bf16* rT   = (__bf16*)grab((size_t)HIDC * INC * 2);
  float* xl    = (float*)grab((size_t)NN * OUTC * 4);
  float* xr    = (float*)grab((size_t)NN * OUTC * 4);
  float* resid = (float*)grab((size_t)NN * HIDC * 4);
  float* hbuf  = (float*)grab((size_t)NN * HIDC * 4);
  float* logits = (float*)grab((size_t)EE * 4 * 4);
  float* exv    = (float*)grab((size_t)EE * 4 * 4);
  unsigned* amaxu = (unsigned*)grab((size_t)NN * 4 * 4);
  float* denom  = (float*)grab((size_t)NN * 4 * 4);
  float* outacc = (float*)grab((size_t)NN * HIDC * 4);

  const int gemmBlocks = CDIV(NN / 16, 8);  // 391

  // x -> bf16; residual projection
  k_tobf<<<CDIV(NN * INC, 256), 256, 0, stream>>>(x, xbf, NN * INC);
  k_wtrans<<<CDIV(INC * HIDC, 256), 256, 0, stream>>>(resW, rT, INC, HIDC);
  k_gemm_node<INC, HIDC><<<gemmBlocks, 256, 0, stream>>>(xbf, rT, resb, resid, NN);

  for (int L = 0; L < 3; ++L) {
    const float *Wl, *bl, *Wr, *br, *We, *attp, *bp;
    int K;
    if (L == 0) {
      Wl = Wl0; bl = bl0; Wr = Wr0; br = br0; We = We0; attp = att0; bp = bias0;
      K = INC;
    } else {
      int i = L - 1;
      Wl = Wl12 + (size_t)i * HIDC * OUTC; bl = bl12 + i * OUTC;
      Wr = Wr12 + (size_t)i * HIDC * OUTC; br = br12 + i * OUTC;
      We = We12 + (size_t)i * INC * OUTC;  attp = att12 + i * OUTC;
      bp = bias12 + i * HIDC;
      K = HIDC;
    }

    k_wtrans<<<CDIV(K * OUTC, 256), 256, 0, stream>>>(Wl, wlT, K, OUTC);
    k_wtrans<<<CDIV(K * OUTC, 256), 256, 0, stream>>>(Wr, wrT, K, OUTC);
    k_wtrans<<<CDIV(INC * OUTC, 256), 256, 0, stream>>>(We, weT, INC, OUTC);

    if (L == 0) {
      k_gemm_node<INC, OUTC><<<gemmBlocks, 256, 0, stream>>>(xbf, wlT, bl, xl, NN);
      k_gemm_node<INC, OUTC><<<gemmBlocks, 256, 0, stream>>>(xbf, wrT, br, xr, NN);
    } else {
      k_gemm_node<HIDC, OUTC><<<gemmBlocks, 256, 0, stream>>>(xbf, wlT, bl, xl, NN);
      k_gemm_node<HIDC, OUTC><<<gemmBlocks, 256, 0, stream>>>(xbf, wrT, br, xr, NN);
    }

    k_zero<<<CDIV(NN * 4, 256), 256, 0, stream>>>(amaxu, NN * 4);
    k_zero<<<CDIV(NN * 4, 256), 256, 0, stream>>>((unsigned*)denom, NN * 4);
    k_zero<<<CDIV(NN * 64, 256), 256, 0, stream>>>((unsigned*)outacc, NN * 64);

    k_edge_logits<<<CDIV(EE, 128), 256, 0, stream>>>(ea, weT, xl, xr, attp, src,
                                                     dst, logits, amaxu);
    k_edge_exp<<<CDIV(EE * 4, 256), 256, 0, stream>>>(logits, amaxu, dst, exv,
                                                      denom);
    k_edge_agg<<<CDIV(EE * 64, 256), 256, 0, stream>>>(exv, denom, xl, src, dst,
                                                       outacc);

    if (L == 0) {
      k_finalize<<<CDIV(NN * 64, 256), 256, 0, stream>>>(
          outacc, bp, bng, bnb, bnm, bnv, resid, hbuf, xbf, nullptr, 0);
    } else if (L == 1) {
      k_finalize<<<CDIV(NN * 64, 256), 256, 0, stream>>>(
          outacc, bp, bng + 64, bnb + 64, bnm + 64, bnv + 64, hbuf, hbuf, xbf,
          nullptr, 0);
    } else {
      k_finalize<<<CDIV(NN * 64, 256), 256, 0, stream>>>(
          outacc, bp, nullptr, nullptr, nullptr, nullptr, nullptr, nullptr,
          nullptr, (float*)d_out, 1);
    }
  }

  // second tuple element: edge_attr passthrough
  k_copyf<<<CDIV(EE * INC, 256), 256, 0, stream>>>(
      ea, (float*)d_out + (size_t)NN * HIDC, EE * INC);
}
